// GraphModel_68436008895089
// MI455X (gfx1250) — matrix operand order repaired and checked by reading, not compiled, not run on hardware
//
#include <hip/hip_runtime.h>
#include <hip/hip_fp16.h>

// ---------------------------------------------------------------------------
// GGNN session-graph recommender forward, CDNA5 (gfx1250, wave32).
// All GEMMs via v_wmma_f32_16x16x32_f16; f32 accumulate; gate math in f32.
// Score stores are non-temporal (TH=NT) to keep the 192MB L2 for operands.
// ---------------------------------------------------------------------------

typedef __attribute__((ext_vector_type(16))) _Float16 v16h;
typedef __attribute__((ext_vector_type(8)))  float    v8f;

#define B_      1024
#define L_      32
#define D_      128
#define NN_     50000

// ---- WMMA fragment loaders (ISA 7.12.2 16-bit layouts, wave32) -------------
// A (16xK row-major, MxK): lane holds row=lane&15; k-pairs at base (lane>>4)*8
// and +16 -> two contiguous 16B chunks per 32-wide k-step.
__device__ __forceinline__ v16h load_a_frag(const _Float16* A, int lda, int lane) {
    int row = lane & 15;
    int kb  = (lane >> 4) << 3;
    const _Float16* p = A + (size_t)row * lda + kb;
    union { v16h v; float4 f[2]; } u;
    u.f[0] = *(const float4*)(p);
    u.f[1] = *(const float4*)(p + 16);
    return u.v;
}
// B (Kx16), sourced from transposed weights Bt[N][K] row-major: lane holds
// col=lane&15; k at base (lane>>4)*8 and +16 -> two contiguous 16B chunks.
__device__ __forceinline__ v16h load_b_frag(const _Float16* Bt, int ldb, int lane) {
    int row = lane & 15;
    int ks  = (lane >> 4) << 3;
    const _Float16* p = Bt + (size_t)row * ldb + ks;
    union { v16h v; float4 f[2]; } u;
    u.f[0] = *(const float4*)(p);
    u.f[1] = *(const float4*)(p + 16);
    return u.v;
}

__device__ __forceinline__ v8f wmma_step(v16h a, v16h b, v8f c) {
    return __builtin_amdgcn_wmma_f32_16x16x32_f16(
        /*neg_a=*/false, a, /*neg_b=*/false, b,
        /*c_mod=*/(short)0, c, /*reuse_a=*/false, /*reuse_b=*/false);
}

// acc += A[16,K] @ Bt[16,K]^T  (A row-major lda, Bt row-major ldb, K % 32 == 0)
__device__ __forceinline__ void gemm_acc(const _Float16* A, int lda,
                                         const _Float16* Bt, int ldb,
                                         int K, int lane, v8f& acc) {
    for (int k0 = 0; k0 < K; k0 += 32) {
        v16h a = load_a_frag(A + k0, lda, lane);
        v16h b = load_b_frag(Bt + k0, ldb, lane);
        acc = wmma_step(a, b, acc);
    }
}

__device__ __forceinline__ float sigmoidf_(float x) { return 1.f / (1.f + __expf(-x)); }

// ---- prep kernels ----------------------------------------------------------
__global__ void k_cvt(_Float16* __restrict__ dst, const float* __restrict__ src, long n) {
    for (long i = (long)blockIdx.x * blockDim.x + threadIdx.x; i < n;
         i += (long)gridDim.x * blockDim.x)
        dst[i] = (_Float16)src[i];
}

// dst[n*K + k] = src[k*N + n]  (f32 -> f16 transpose)
__global__ void k_tcvt(_Float16* __restrict__ dst, const float* __restrict__ src,
                       int K, int N) {
    long tot = (long)K * N;
    for (long i = (long)blockIdx.x * blockDim.x + threadIdx.x; i < tot;
         i += (long)gridDim.x * blockDim.x) {
        int n = (int)(i / K);
        int k = (int)(i - (long)n * K);
        dst[i] = (_Float16)src[(long)k * N + n];
    }
}

// h16[i, d] = emb[items[i], d]
__global__ void k_gather(_Float16* __restrict__ h16, const float* __restrict__ emb,
                         const int* __restrict__ items, int rows) {
    long tot = (long)rows * D_;
    for (long idx = (long)blockIdx.x * blockDim.x + threadIdx.x; idx < tot;
         idx += (long)gridDim.x * blockDim.x) {
        long i = idx >> 7;
        int  d = (int)(idx & 127);
        h16[idx] = (_Float16)emb[(size_t)items[i] * D_ + d];
    }
}

// ---- K1: hin/hout = h @ W_{in,out} + b, stored TRANSPOSED per batch --------
// hinT layout: [B][D][L]  (B-operand for the adjacency GEMM)
__global__ void k1_project(const _Float16* __restrict__ h16,
                           const _Float16* __restrict__ WtIn,  const float* __restrict__ bIn,
                           const _Float16* __restrict__ WtOut, const float* __restrict__ bOut,
                           _Float16* __restrict__ hinT, _Float16* __restrict__ houtT) {
    int lane = threadIdx.x & 31;
    int w    = threadIdx.x >> 5;      // 8 waves -> 8 N-tiles
    int m0   = blockIdx.x * 16;
    int n0   = w * 16;
    const _Float16* A = h16 + (size_t)m0 * D_;
    for (int p = 0; p < 2; ++p) {
        const _Float16* Bt  = (p ? WtOut : WtIn) + (size_t)n0 * D_;
        const float*   bias = p ? bOut : bIn;
        _Float16*      dst  = p ? houtT : hinT;
        v8f acc = {};
        gemm_acc(A, D_, Bt, D_, D_, lane, acc);
        int hi8 = (lane >> 4) << 3;
        int n   = n0 + (lane & 15);
        float bv = bias[n];
        for (int r = 0; r < 8; ++r) {
            int m  = m0 + r + hi8;
            int bb = m >> 5, rr = m & 31;
            dst[((size_t)bb * D_ + n) * L_ + rr] = (_Float16)(acc[r] + bv);
        }
    }
}

// ---- K2: a_in/a_out = A_{in,out}[32x32] @ hin[32x128]; concat into acat ----
__global__ void k2_adj(const _Float16* __restrict__ A16in, const _Float16* __restrict__ A16out,
                       const _Float16* __restrict__ hinT,  const _Float16* __restrict__ houtT,
                       _Float16* __restrict__ acat) {
    int lane = threadIdx.x & 31;
    int w    = threadIdx.x >> 5;
    int b    = blockIdx.x;
    for (int j = 0; j < 4; ++j) {
        int tid   = w * 4 + j;          // 0..31 tiles
        int which = tid >> 4;           // 0 = in, 1 = out
        int tt    = tid & 15;
        int m0    = (tt & 1) * 16;
        int n0    = (tt >> 1) * 16;
        const _Float16* A  = (which ? A16out : A16in) + (size_t)b * (L_ * L_) + (size_t)m0 * L_;
        const _Float16* Bt = (which ? houtT : hinT)   + (size_t)b * (D_ * L_) + (size_t)n0 * L_;
        v8f acc = {};
        gemm_acc(A, L_, Bt, L_, L_, lane, acc);     // K = 32, single wmma step
        int hi8 = (lane >> 4) << 3;
        int n   = n0 + (lane & 15);
        for (int r = 0; r < 8; ++r) {
            int m = m0 + r + hi8;
            acat[((size_t)b * L_ + m) * (2 * D_) + which * D_ + n] = (_Float16)acc[r];
        }
    }
}

// ---- fused GRU: out = gru(A1@Bt1'+bias1, A2@Bt2'+bias2, Hprev) -------------
// stream1 -> gi (r,z,n), stream2 -> gh. r,z: sigmoid(s1+s2). n: tanh(s1 + r*s2).
__global__ void k_gru(const _Float16* __restrict__ A1, const _Float16* __restrict__ Bt1,
                      int K1, const float* __restrict__ bias1,
                      const _Float16* __restrict__ A2, const _Float16* __restrict__ Bt2,
                      int K2, const float* __restrict__ bias2,
                      const _Float16* __restrict__ Hprev,
                      _Float16* __restrict__ out16, float* __restrict__ out32) {
    __shared__ float S1[16 * 384];   // gi(+bias) for all 384; for n<256 also += gh
    __shared__ float S2[16 * 128];   // gh n-part (h_n)
    int lane = threadIdx.x & 31;
    int w    = threadIdx.x >> 5;
    int m0   = blockIdx.x * 16;
    for (int j = 0; j < 3; ++j) {
        int t  = w * 3 + j;          // 24 N-tiles over N=384
        int n0 = t * 16;
        int hi8 = (lane >> 4) << 3;
        int nl  = n0 + (lane & 15);
        v8f acc = {};
        gemm_acc(A1 + (size_t)m0 * K1, K1, Bt1 + (size_t)n0 * K1, K1, K1, lane, acc);
        if (t < 16) {                // r,z region: sum both streams
            gemm_acc(A2 + (size_t)m0 * K2, K2, Bt2 + (size_t)n0 * K2, K2, K2, lane, acc);
            float bv = (bias1 ? bias1[nl] : 0.f) + (bias2 ? bias2[nl] : 0.f);
            for (int r = 0; r < 8; ++r) S1[(r + hi8) * 384 + nl] = acc[r] + bv;
        } else {                     // n region: keep streams separate
            float bv = bias1 ? bias1[nl] : 0.f;
            for (int r = 0; r < 8; ++r) S1[(r + hi8) * 384 + nl] = acc[r] + bv;
            v8f acc2 = {};
            gemm_acc(A2 + (size_t)m0 * K2, K2, Bt2 + (size_t)n0 * K2, K2, K2, lane, acc2);
            float bv2 = bias2 ? bias2[nl] : 0.f;
            for (int r = 0; r < 8; ++r) S2[(r + hi8) * 128 + (nl - 256)] = acc2[r] + bv2;
        }
    }
    __syncthreads();
    for (int i = 0; i < 8; ++i) {
        int idx = threadIdx.x + i * 256;   // 16 rows x 128 d
        int m = idx >> 7, d = idx & 127;
        float r  = sigmoidf_(S1[m * 384 + d]);
        float z  = sigmoidf_(S1[m * 384 + 128 + d]);
        float nn = tanhf(S1[m * 384 + 256 + d] + r * S2[m * 128 + d]);
        float hp = (float)Hprev[(size_t)(m0 + m) * D_ + d];
        float o  = (1.f - z) * nn + z * hp;
        out16[(size_t)(m0 + m) * D_ + d] = (_Float16)o;
        if (out32) out32[(size_t)(m0 + m) * D_ + d] = o;
    }
}

// ---- K5: attention readout, one block per batch ----------------------------
__global__ void k5_attn(const _Float16* __restrict__ final16, const float* __restrict__ final32,
                        const int* __restrict__ seq_len,
                        const float* __restrict__ W1, const float* __restrict__ b1,
                        const _Float16* __restrict__ Wt2, const float* __restrict__ b2,
                        const float* __restrict__ wq, const float* __restrict__ bq,
                        const float* __restrict__ W3, const float* __restrict__ b3,
                        _Float16* __restrict__ hs16) {
    __shared__ float FW[32 * 128];
    __shared__ float vn[128];
    __shared__ float vw[128];
    __shared__ float alpha[32];
    __shared__ float sg[128];
    int b    = blockIdx.x;
    int tid  = threadIdx.x;
    int lane = tid & 31;
    int w    = tid >> 5;
    int sl   = seq_len[b];
    if (tid < 128) vn[tid] = final32[((size_t)b * L_ + (sl - 1)) * D_ + tid];
    // FW = final[b] @ W2  (16 tiles, 2 per wave)
    for (int j = 0; j < 2; ++j) {
        int t  = w * 2 + j;
        int m0 = (t & 1) * 16;
        int n0 = (t >> 1) * 16;
        const _Float16* A  = final16 + ((size_t)b * L_ + m0) * D_;
        const _Float16* Bt = Wt2 + (size_t)n0 * D_;
        v8f acc = {};
        gemm_acc(A, D_, Bt, D_, D_, lane, acc);
        int hi8 = (lane >> 4) << 3;
        int n   = n0 + (lane & 15);
        for (int r = 0; r < 8; ++r) FW[(m0 + r + hi8) * 128 + n] = acc[r];
    }
    __syncthreads();
    if (tid < 128) {                  // v_n @ W1 + b1 + b2
        float s = 0.f;
        for (int k = 0; k < 128; ++k) s += vn[k] * W1[k * 128 + tid];
        vw[tid] = s + b1[tid] + b2[tid];
    }
    __syncthreads();
    if (tid < 32) {                   // alpha = sigmoid(...)@wq + bq, masked
        float a = 0.f;
        for (int d = 0; d < 128; ++d) a += sigmoidf_(vw[d] + FW[tid * 128 + d]) * wq[d];
        a += bq[0];
        alpha[tid] = (tid < sl) ? a : 0.f;
    }
    __syncthreads();
    if (tid < 128) {                  // s_g = sum_l alpha_l * final_l
        float s = 0.f;
        for (int l = 0; l < 32; ++l) s += alpha[l] * final32[((size_t)b * L_ + l) * D_ + tid];
        sg[tid] = s;
    }
    __syncthreads();
    if (tid < 128) {                  // h_s = [v_n, s_g] @ W3 + b3
        float acc = b3[tid];
        for (int k = 0; k < 128; ++k) acc += vn[k] * W3[k * 128 + tid];
        for (int k = 0; k < 128; ++k) acc += sg[k] * W3[(128 + k) * 128 + tid];
        hs16[(size_t)b * 128 + tid] = (_Float16)acc;
    }
}

// ---- K6: scores = h_s @ emb.T  (M=1024, N=50000, K=128) --------------------
// Output is write-once, never re-read: use non-temporal stores so the 205MB
// score stream doesn't evict emb16/hs16 from the 192MB L2.
__global__ void k6_scores(const _Float16* __restrict__ hs16,
                          const _Float16* __restrict__ emb16,
                          float* __restrict__ out) {
    int lane  = threadIdx.x & 31;
    int w     = threadIdx.x >> 5;
    int m0    = (blockIdx.y * 4 + w) * 16;
    int n0    = blockIdx.x * 16;
    v8f acc = {};
    gemm_acc(hs16 + (size_t)m0 * D_, D_, emb16 + (size_t)n0 * D_, D_, D_, lane, acc);
    int hi8 = (lane >> 4) << 3;
    int n   = n0 + (lane & 15);
    for (int r = 0; r < 8; ++r) {
        int m = m0 + r + hi8;
        __builtin_nontemporal_store(acc[r], &out[(size_t)m * NN_ + n]);
    }
}

// ---------------------------------------------------------------------------
extern "C" void kernel_launch(void* const* d_in, const int* in_sizes, int n_in,
                              void* d_out, int out_size, void* d_ws, size_t ws_size,
                              hipStream_t stream) {
    const int*   items   = (const int*)d_in[0];
    const float* A_in    = (const float*)d_in[1];
    const float* A_out   = (const float*)d_in[2];
    const float* inter   = (const float*)d_in[3];
    const int*   seq_len = (const int*)d_in[4];
    const float* emb     = (const float*)d_in[5];
    const float* W_in    = (const float*)d_in[6];
    const float* b_in    = (const float*)d_in[7];
    const float* W_out   = (const float*)d_in[8];
    const float* b_out   = (const float*)d_in[9];
    const float* W_a     = (const float*)d_in[10];
    const float* U_h     = (const float*)d_in[11];
    const float* b_gru   = (const float*)d_in[12];
    const float* Wi      = (const float*)d_in[13];
    const float* bi      = (const float*)d_in[14];
    const float* Wh      = (const float*)d_in[15];
    const float* bh      = (const float*)d_in[16];
    const float* W1      = (const float*)d_in[17];
    const float* b1      = (const float*)d_in[18];
    const float* W2      = (const float*)d_in[19];
    const float* b2      = (const float*)d_in[20];
    const float* wq      = (const float*)d_in[21];
    const float* bq      = (const float*)d_in[22];
    const float* W3      = (const float*)d_in[23];
    const float* b3      = (const float*)d_in[24];
    float* scores = (float*)d_out;

    char* ws = (char*)d_ws;
    size_t off = 0;
    auto alloc = [&](size_t bytes) -> char* {
        char* p = ws + off;
        off = (off + bytes + 255) & ~(size_t)255;
        return p;
    };
    const size_t BL = (size_t)B_ * L_;
    _Float16* emb16  = (_Float16*)alloc((size_t)NN_ * D_ * 2);
    _Float16* h16    = (_Float16*)alloc(BL * D_ * 2);
    _Float16* hinT   = (_Float16*)alloc((size_t)B_ * D_ * L_ * 2);
    _Float16* houtT  = (_Float16*)alloc((size_t)B_ * D_ * L_ * 2);
    _Float16* acat   = (_Float16*)alloc(BL * 2 * D_ * 2);     // 16.8 MB
    _Float16* intra16= (_Float16*)alloc(BL * D_ * 2);
    _Float16* inter16= (_Float16*)alloc(BL * D_ * 2);
    _Float16* A16in  = (_Float16*)alloc(BL * L_ * 2);
    _Float16* A16out = (_Float16*)alloc(BL * L_ * 2);
    _Float16* WtIn   = (_Float16*)alloc((size_t)D_ * D_ * 2);
    _Float16* WtOut  = (_Float16*)alloc((size_t)D_ * D_ * 2);
    _Float16* WtA    = (_Float16*)alloc((size_t)256 * 384 * 2);
    _Float16* UtH    = (_Float16*)alloc((size_t)128 * 384 * 2);
    _Float16* WtI    = (_Float16*)alloc((size_t)128 * 384 * 2);
    _Float16* WtH    = (_Float16*)alloc((size_t)128 * 384 * 2);
    _Float16* Wt2    = (_Float16*)alloc((size_t)D_ * D_ * 2);
    _Float16* hs16   = (_Float16*)alloc((size_t)B_ * D_ * 2);
    // aliases: dead buffers reused (sizes match exactly)
    float*    final32 = (float*)acat;        // acat dead after k3
    _Float16* final16 = hinT;                // hinT dead after k2

    dim3 cb(256);
    // conversions / transposes / gather
    k_cvt   <<<dim3(1024), cb, 0, stream>>>(emb16,  emb,   (long)NN_ * D_);
    k_cvt   <<<dim3(512),  cb, 0, stream>>>(A16in,  A_in,  (long)BL * L_);
    k_cvt   <<<dim3(512),  cb, 0, stream>>>(A16out, A_out, (long)BL * L_);
    k_cvt   <<<dim3(1024), cb, 0, stream>>>(inter16, inter, (long)BL * D_);
    k_gather<<<dim3(1024), cb, 0, stream>>>(h16, emb, items, (int)BL);
    k_tcvt  <<<dim3(64),  cb, 0, stream>>>(WtIn,  W_in,  128, 128);
    k_tcvt  <<<dim3(64),  cb, 0, stream>>>(WtOut, W_out, 128, 128);
    k_tcvt  <<<dim3(384), cb, 0, stream>>>(WtA,   W_a,   256, 384);
    k_tcvt  <<<dim3(192), cb, 0, stream>>>(UtH,   U_h,   128, 384);
    k_tcvt  <<<dim3(192), cb, 0, stream>>>(WtI,   Wi,    128, 384);
    k_tcvt  <<<dim3(192), cb, 0, stream>>>(WtH,   Wh,    128, 384);
    k_tcvt  <<<dim3(64),  cb, 0, stream>>>(Wt2,   W2,    128, 128);

    // K1: hin/hout projections (transposed per batch)
    k1_project<<<dim3(BL / 16), cb, 0, stream>>>(h16, WtIn, b_in, WtOut, b_out, hinT, houtT);
    // K2: adjacency matmuls -> acat = [a_in | a_out]
    k2_adj<<<dim3(B_), cb, 0, stream>>>(A16in, A16out, hinT, houtT, acat);
    // K3: intra = GRU(acat@W_a + b_gru, h@U_h, h)
    k_gru<<<dim3(BL / 16), cb, 0, stream>>>(acat, WtA, 256, b_gru,
                                            h16, UtH, 128, (const float*)nullptr,
                                            h16, intra16, (float*)nullptr);
    // K4: final = GRU(intra@Wi + bi, inter@Wh + bh, inter)
    k_gru<<<dim3(BL / 16), cb, 0, stream>>>(intra16, WtI, 128, bi,
                                            inter16, WtH, 128, bh,
                                            inter16, final16, final32);
    // K5: attention readout -> h_s (f16)
    k5_attn<<<dim3(B_), cb, 0, stream>>>(final16, final32, seq_len,
                                         W1, b1, Wt2, b2, wq, bq, W3, b3, hs16);
    // K6: scores = h_s @ emb.T  (writes all of d_out, non-temporal)
    k6_scores<<<dim3(NN_ / 16, B_ / 16 / 4), dim3(128), 0, stream>>>(hs16, emb16, scores);
}
